// GraphConv_15590731285087
// MI455X (gfx1250) — matrix operand order repaired and checked by reading, not compiled, not run on hardware
//
#include <hip/hip_runtime.h>
#include <hip/hip_bf16.h>

#define D 128
#define HST 136       // padded LDS row stride (bf16 elems) for inter-layer staging
#define ROW_ITERS 4   // row-tiles per wave; 512 rows per workgroup

typedef __attribute__((ext_vector_type(16))) __bf16 v16bf;
typedef __attribute__((ext_vector_type(4)))  __bf16 v4bf;
typedef __attribute__((ext_vector_type(8)))  float  v8f;

// Load a 32x16 B fragment stored in fragment order (32 lanes x 16 contiguous bf16).
static __device__ __forceinline__ v16bf load_b_frag(const __bf16* w, int lane) {
  return *(const v16bf*)(w + lane * 16);
}

// ---------------------------------------------------------------------------
// Convert fp32 weight [K][128] (row-major, x @ W) into bf16 B-fragment order:
// tile t = nt*(K/32)+kt ; within tile: lane l holds k = kt*32 + (e>>3)*16 + (l>>4)*8 + (e&7),
// n = nt*16 + (l&15) for e = 0..15 (matches V_WMMA 16-bit B layout).
// ---------------------------------------------------------------------------
__global__ void prep_weights(const float* __restrict__ src,
                             __bf16* __restrict__ dst, int K) {
  int i = blockIdx.x * blockDim.x + threadIdx.x;
  if (i >= K * D) return;
  int k = i / D, n = i % D;
  int kt = k >> 5, kk = k & 31;
  int nt = n >> 4, nl = n & 15;
  int l = (((kk >> 3) & 1) << 4) | nl;
  int e = ((kk >> 4) << 3) | (kk & 7);
  int t = nt * (K >> 5) + kt;
  dst[t * 512 + l * 16 + e] = (__bf16)src[i];
}

__global__ void zero_f32(float4* __restrict__ p, long n4) {
  long i = (long)blockIdx.x * blockDim.x + threadIdx.x;
  long stride = (long)gridDim.x * blockDim.x;
  float4 z; z.x = 0.f; z.y = 0.f; z.z = 0.f; z.w = 0.f;
  for (; i < n4; i += stride) p[i] = z;
}

// Build an A fragment (16x32 bf16) from a row-major fp32 matrix.
// Lane layout: m = lane&15, kb = (lane>>4)*8; elems 0-7 -> k=kb..kb+7,
// elems 8-15 -> k=kb+16..kb+23 (per ISA 16-bit A-matrix table).
static __device__ __forceinline__ v16bf load_a_f32(const float* xp, bool valid) {
  v16bf u;
  if (valid) {
    const float4* q = (const float4*)xp;
    float4 x0 = q[0], x1 = q[1], x2 = q[4], x3 = q[5];
    u[0]  = (__bf16)x0.x; u[1]  = (__bf16)x0.y; u[2]  = (__bf16)x0.z; u[3]  = (__bf16)x0.w;
    u[4]  = (__bf16)x1.x; u[5]  = (__bf16)x1.y; u[6]  = (__bf16)x1.z; u[7]  = (__bf16)x1.w;
    u[8]  = (__bf16)x2.x; u[9]  = (__bf16)x2.y; u[10] = (__bf16)x2.z; u[11] = (__bf16)x2.w;
    u[12] = (__bf16)x3.x; u[13] = (__bf16)x3.y; u[14] = (__bf16)x3.z; u[15] = (__bf16)x3.w;
  } else {
#pragma unroll
    for (int i = 0; i < 16; ++i) u[i] = (__bf16)0.f;
  }
  return u;
}

static __device__ __forceinline__ v16bf load_a_lds(const __bf16* hp) {
  v16bf u;
#pragma unroll
  for (int i = 0; i < 8; ++i) u[i] = hp[i];
#pragma unroll
  for (int i = 0; i < 8; ++i) u[8 + i] = hp[16 + i];
  return u;
}

// ---------------------------------------------------------------------------
// Message MLP: Z = relu(relu(X @ W1 + b1) @ W2 + b2), X:[N,128] fp32, Z bf16.
// grid.y selects the MLP (pos/neg). 8 waves/block, 16 rows/wave/iter,
// ROW_ITERS iterations per block to amortize the LDS-resident weights.
// ---------------------------------------------------------------------------
__global__ void __launch_bounds__(256) mlp_msg(
    const float* __restrict__ X, const float* __restrict__ b1base,
    const float* __restrict__ b2base, const __bf16* __restrict__ w1base,
    const __bf16* __restrict__ w2base, __bf16* __restrict__ Zbase,
    int N) {
  extern __shared__ __bf16 smem[];
  __bf16* sW1 = smem;          // 16384 (32 KB)
  __bf16* sW2 = smem + 16384;  // 16384 (32 KB)
  __bf16* sH  = smem + 32768;  // 8 waves * 16 * HST

  const int mlp = blockIdx.y;
  const __bf16* gW1 = w1base + mlp * 16384;
  const __bf16* gW2 = w2base + mlp * 16384;
  const float* b1 = b1base + mlp * D;
  const float* b2 = b2base + mlp * D;
  __bf16* Z = Zbase + (size_t)mlp * N * D;

  {  // stage both weight layers in LDS
    const uint4* s1 = (const uint4*)gW1;
    const uint4* s2 = (const uint4*)gW2;
    uint4* d1 = (uint4*)sW1;
    uint4* d2 = (uint4*)sW2;
    for (int i = threadIdx.x; i < 2048; i += blockDim.x) { d1[i] = s1[i]; d2[i] = s2[i]; }
  }
  __syncthreads();

  const int wave = threadIdx.x >> 5;
  const int lane = threadIdx.x & 31;
  const int nl   = lane & 15;   // A row within tile == C/D column within tile
  const int lhi  = lane >> 4;
  __bf16* H = sH + wave * 16 * HST;

  for (int it = 0; it < ROW_ITERS; ++it) {
    const long row0 = ((long)blockIdx.x * ROW_ITERS + it) * 128 + wave * 16;
    if (row0 >= N) break;
    const long arow = row0 + nl;
    const bool avalid = arow < N;

    // ---- layer 1: A from global fp32, B from LDS, acc fp32 ----
    v16bf a[4];
    const float* xp = X + (size_t)(avalid ? arow : 0) * D + lhi * 8;
#pragma unroll
    for (int kt = 0; kt < 4; ++kt) a[kt] = load_a_f32(xp + kt * 32, avalid);

#pragma unroll
    for (int nt = 0; nt < 8; ++nt) {
      v8f acc = {};
#pragma unroll
      for (int kt = 0; kt < 4; ++kt) {
        v16bf b = load_b_frag(sW1 + (nt * 4 + kt) * 512, lane);
        acc = __builtin_amdgcn_wmma_f32_16x16x32_bf16(false, a[kt], false, b,
                                                      (short)0, acc, false, false);
      }
      float bias = b1[nt * 16 + nl];
#pragma unroll
      for (int i = 0; i < 8; ++i) {
        float v = fmaxf(acc[i] + bias, 0.f);
        H[(i + 8 * lhi) * HST + nt * 16 + nl] = (__bf16)v;
      }
    }

    // ---- layer 2: A from LDS staging ----
    v16bf a2[4];
#pragma unroll
    for (int kt = 0; kt < 4; ++kt)
      a2[kt] = load_a_lds(H + nl * HST + kt * 32 + lhi * 8);

#pragma unroll
    for (int nt = 0; nt < 8; ++nt) {
      v8f acc = {};
#pragma unroll
      for (int kt = 0; kt < 4; ++kt) {
        v16bf b = load_b_frag(sW2 + (nt * 4 + kt) * 512, lane);
        acc = __builtin_amdgcn_wmma_f32_16x16x32_bf16(false, a2[kt], false, b,
                                                      (short)0, acc, false, false);
      }
      float bias = b2[nt * 16 + nl];
#pragma unroll
      for (int i = 0; i < 8; ++i) {
        long r = row0 + i + 8 * lhi;
        if (r < N) {
          float v = fmaxf(acc[i] + bias, 0.f);
          Z[(size_t)r * D + nt * 16 + nl] = (__bf16)v;
        }
      }
    }
  }
}

// ---------------------------------------------------------------------------
// Update MLP: Out = relu(relu(concat(H,M) @ W1 + b1) @ W2 + b2), fp32 out.
// ---------------------------------------------------------------------------
__global__ void __launch_bounds__(256) mlp_upd(
    const float* __restrict__ Hin, const float* __restrict__ M,
    const float* __restrict__ b1, const float* __restrict__ b2,
    const __bf16* __restrict__ gW1, const __bf16* __restrict__ gW2,
    float* __restrict__ Out, int N) {
  extern __shared__ __bf16 smem[];
  __bf16* sW1 = smem;          // 32768 (64 KB, K=256)
  __bf16* sW2 = smem + 32768;  // 16384 (32 KB)
  __bf16* sH  = smem + 49152;

  {
    const uint4* s1 = (const uint4*)gW1;
    uint4* d1 = (uint4*)sW1;
    for (int i = threadIdx.x; i < 4096; i += blockDim.x) d1[i] = s1[i];
    const uint4* s2 = (const uint4*)gW2;
    uint4* d2 = (uint4*)sW2;
    for (int i = threadIdx.x; i < 2048; i += blockDim.x) d2[i] = s2[i];
  }
  __syncthreads();

  const int wave = threadIdx.x >> 5;
  const int lane = threadIdx.x & 31;
  const int nl   = lane & 15;
  const int lhi  = lane >> 4;
  __bf16* H = sH + wave * 16 * HST;

  for (int it = 0; it < ROW_ITERS; ++it) {
    const long row0 = ((long)blockIdx.x * ROW_ITERS + it) * 128 + wave * 16;
    if (row0 >= N) break;
    const long arow = row0 + nl;
    const bool avalid = arow < N;

    // ---- layer 1: K=256, cols 0-127 from Hin, 128-255 from M ----
    v16bf a[8];
    {
      const float* hp = Hin + (size_t)(avalid ? arow : 0) * D + lhi * 8;
      const float* mp = M   + (size_t)(avalid ? arow : 0) * D + lhi * 8;
#pragma unroll
      for (int kt = 0; kt < 4; ++kt) a[kt]     = load_a_f32(hp + kt * 32, avalid);
#pragma unroll
      for (int kt = 0; kt < 4; ++kt) a[kt + 4] = load_a_f32(mp + kt * 32, avalid);
    }

#pragma unroll
    for (int nt = 0; nt < 8; ++nt) {
      v8f acc = {};
#pragma unroll
      for (int kt = 0; kt < 8; ++kt) {
        v16bf b = load_b_frag(sW1 + (nt * 8 + kt) * 512, lane);
        acc = __builtin_amdgcn_wmma_f32_16x16x32_bf16(false, a[kt], false, b,
                                                      (short)0, acc, false, false);
      }
      float bias = b1[nt * 16 + nl];
#pragma unroll
      for (int i = 0; i < 8; ++i) {
        float v = fmaxf(acc[i] + bias, 0.f);
        H[(i + 8 * lhi) * HST + nt * 16 + nl] = (__bf16)v;
      }
    }

    // ---- layer 2 ----
    v16bf a2[4];
#pragma unroll
    for (int kt = 0; kt < 4; ++kt)
      a2[kt] = load_a_lds(H + nl * HST + kt * 32 + lhi * 8);

#pragma unroll
    for (int nt = 0; nt < 8; ++nt) {
      v8f acc = {};
#pragma unroll
      for (int kt = 0; kt < 4; ++kt) {
        v16bf b = load_b_frag(sW2 + (nt * 4 + kt) * 512, lane);
        acc = __builtin_amdgcn_wmma_f32_16x16x32_bf16(false, a2[kt], false, b,
                                                      (short)0, acc, false, false);
      }
      float bias = b2[nt * 16 + nl];
#pragma unroll
      for (int i = 0; i < 8; ++i) {
        long r = row0 + i + 8 * lhi;
        if (r < N) {
          Out[(size_t)r * D + nt * 16 + nl] = fmaxf(acc[i] + bias, 0.f);
        }
      }
    }
  }
}

// ---------------------------------------------------------------------------
// Edge scatter-add: Mout[row[e]] += Z[col[e]]  (bf16 gather, f32 HW atomics).
// One wave per edge: 4 elements per lane.
// ---------------------------------------------------------------------------
__global__ void scatter_add(const __bf16* __restrict__ Zt,
                            const int* __restrict__ col,
                            const int* __restrict__ row,
                            float* __restrict__ Mout, int E) {
  int gtid = blockIdx.x * blockDim.x + threadIdx.x;
  int wid = gtid >> 5;
  int lane = gtid & 31;
  int nw = (gridDim.x * blockDim.x) >> 5;
  for (int e = wid; e < E; e += nw) {
    int c = col[e];
    int r = row[e];
    const __bf16* zp = Zt + (size_t)c * D + lane * 4;
    uint2 packed = *(const uint2*)zp;
    v4bf z = __builtin_bit_cast(v4bf, packed);
    float* op = Mout + (size_t)r * D + lane * 4;
    __hip_atomic_fetch_add(op + 0, (float)z[0], __ATOMIC_RELAXED, __HIP_MEMORY_SCOPE_AGENT);
    __hip_atomic_fetch_add(op + 1, (float)z[1], __ATOMIC_RELAXED, __HIP_MEMORY_SCOPE_AGENT);
    __hip_atomic_fetch_add(op + 2, (float)z[2], __ATOMIC_RELAXED, __HIP_MEMORY_SCOPE_AGENT);
    __hip_atomic_fetch_add(op + 3, (float)z[3], __ATOMIC_RELAXED, __HIP_MEMORY_SCOPE_AGENT);
  }
}

extern "C" void kernel_launch(void* const* d_in, const int* in_sizes, int n_in,
                              void* d_out, int out_size, void* d_ws, size_t ws_size,
                              hipStream_t stream) {
  const float* hv = (const float*)d_in[0];
  const float* hc = (const float*)d_in[1];
  const int* row_v = (const int*)d_in[2];
  const int* col_v = (const int*)d_in[3];
  const int* row_c = (const int*)d_in[4];
  const int* col_c = (const int*)d_in[5];
  const float* mw1 = (const float*)d_in[6];
  const float* mb1 = (const float*)d_in[7];
  const float* mw2 = (const float*)d_in[8];
  const float* mb2 = (const float*)d_in[9];
  const float* uw1 = (const float*)d_in[10];
  const float* ub1 = (const float*)d_in[11];
  const float* uw2 = (const float*)d_in[12];
  const float* ub2 = (const float*)d_in[13];

  const int NV = in_sizes[0] / D;
  const int NC = in_sizes[1] / D;
  const int E  = in_sizes[2];
  const int RPB = 128 * ROW_ITERS;  // rows per block

  char* ws = (char*)d_ws;
  size_t off = 0;
  auto alloc = [&](size_t bytes) -> void* {
    off = (off + 255) & ~(size_t)255;
    void* p = ws + off;
    off += bytes;
    return p;
  };
  __bf16* wm1 = (__bf16*)alloc((size_t)4 * 16384 * 2);
  __bf16* wm2 = (__bf16*)alloc((size_t)4 * 16384 * 2);
  __bf16* wu1 = (__bf16*)alloc((size_t)2 * 32768 * 2);
  __bf16* wu2 = (__bf16*)alloc((size_t)2 * 16384 * 2);
  __bf16* zv  = (__bf16*)alloc((size_t)2 * NC * D * 2);
  __bf16* zc  = (__bf16*)alloc((size_t)2 * NV * D * 2);
  float* mv = (float*)alloc((size_t)NV * D * 4);
  float* mc = (float*)alloc((size_t)NC * D * 4);

  // 1) weights -> bf16 fragment order
  for (int i = 0; i < 4; ++i) {
    prep_weights<<<(128 * D + 255) / 256, 256, 0, stream>>>(mw1 + (size_t)i * 128 * D, wm1 + i * 16384, 128);
    prep_weights<<<(128 * D + 255) / 256, 256, 0, stream>>>(mw2 + (size_t)i * 128 * D, wm2 + i * 16384, 128);
  }
  for (int i = 0; i < 2; ++i) {
    prep_weights<<<(256 * D + 255) / 256, 256, 0, stream>>>(uw1 + (size_t)i * 256 * D, wu1 + i * 32768, 256);
    prep_weights<<<(128 * D + 255) / 256, 256, 0, stream>>>(uw2 + (size_t)i * 128 * D, wu2 + i * 16384, 128);
  }

  // 2) zero message accumulators
  zero_f32<<<2048, 256, 0, stream>>>((float4*)mv, (long)NV * D / 4);
  zero_f32<<<2048, 256, 0, stream>>>((float4*)mc, (long)NC * D / 4);

  // 3) message MLPs (grid.y = pos/neg MLP)
  const size_t msg_lds = (size_t)(16384 + 16384 + 8 * 16 * HST) * 2;  // 100352 B
  dim3 gzv((NC + RPB - 1) / RPB, 2);
  mlp_msg<<<gzv, 256, msg_lds, stream>>>(hc, mb1, mb2, wm1, wm2, zv, NC);
  dim3 gzc((NV + RPB - 1) / RPB, 2);
  mlp_msg<<<gzc, 256, msg_lds, stream>>>(hv, mb1 + 2 * D, mb2 + 2 * D,
                                         wm1 + 2 * 16384, wm2 + 2 * 16384, zc, NV);

  // 4) edge scatter-adds
  scatter_add<<<4096, 256, 0, stream>>>(zv, col_v, row_v, mv, E);
  scatter_add<<<4096, 256, 0, stream>>>(zc, col_c, row_c, mc, E);

  // 5) update MLPs -> d_out (hv_new then hc_new)
  float* out = (float*)d_out;
  const size_t upd_lds = (size_t)(32768 + 16384 + 8 * 16 * HST) * 2;  // 133120 B
  mlp_upd<<<(NV + RPB - 1) / RPB, 256, upd_lds, stream>>>(hv, mv, ub1, ub2, wu1, wu2, out, NV);
  mlp_upd<<<(NC + RPB - 1) / RPB, 256, upd_lds, stream>>>(hc, mc, ub1 + D, ub2 + D,
                                                          wu1 + 32768, wu2 + 16384,
                                                          out + (size_t)NV * D, NC);
}
